// WeatherPrediction_29137058136093
// MI455X (gfx1250) — matrix operand order repaired and checked by reading, not compiled, not run on hardware
//
#include <hip/hip_runtime.h>
#include <hip/hip_bf16.h>
#include <stdint.h>
#include <stddef.h>

// ---------------- problem constants ----------------
#define NGRID 65160
#define NMESH 10242
#define NTOT  75402
#define EENC  260640
#define EPROC 81920
#define EDEC  195480
#define NTPAD 75520   // ceil(75402/128)*128
#define NMPAD 10368   // ceil(10242/128)*128

typedef unsigned short ushort_t;

// ---------------- WMMA types ----------------
typedef __attribute__((ext_vector_type(16))) __bf16 v16bf;
typedef __attribute__((ext_vector_type(8)))  float  v8f;

union BFV { v16bf v; unsigned short u[16]; };

__device__ __forceinline__ unsigned short f32_to_bf16_rne(float x) {
  union { float f; unsigned u; } a; a.f = x;
  unsigned r = a.u + 0x7FFFu + ((a.u >> 16) & 1u);
  return (unsigned short)(r >> 16);
}
__device__ __forceinline__ float bf16_bits_to_f32(unsigned short h) {
  union { unsigned u; float f; } a; a.u = ((unsigned)h) << 16;
  return a.f;
}
// split fp32 into bf16 hi + bf16 residual lo  (x ~= hi + lo, rel err ~2^-16)
__device__ __forceinline__ void split2(float x, unsigned short& hi, unsigned short& lo) {
  unsigned short h = f32_to_bf16_rne(x);
  hi = h;
  lo = f32_to_bf16_rne(x - bf16_bits_to_f32(h));
}

__device__ __forceinline__ void atomic_add_f32(float* p, float v) {
  __hip_atomic_fetch_add(p, v, __ATOMIC_RELAXED, __HIP_MEMORY_SCOPE_AGENT);
}

// CDNA5 async global->LDS copy of one 16B chunk per lane (tracked by ASYNCcnt)
__device__ __forceinline__ void async_copy16(unsigned lds_off, const void* gptr) {
  unsigned long long ga = (unsigned long long)gptr;
  asm volatile("global_load_async_to_lds_b128 %0, %1, off"
               :: "v"(lds_off), "v"(ga) : "memory");
}
__device__ __forceinline__ void wait_async0() {
  asm volatile("s_wait_asynccnt 0x0" ::: "memory");
}

// ---------------- pre-split bf16 WMMA GEMM with async LDS staging ----------------
// C[M,N] = (Ahi+Alo)[M,Kpad] @ (Whi+Wlo)[Kpad,Npad] (+bias), operands pre-split
// to bf16 hi/lo and zero-padded to tile multiples (rows of A to 128, K to 32,
// N to 128) so the inner loop has no bounds checks and no conversion math.
// Block: 128x128 tile, 8 waves; wave owns 32x64 = 2x4 fragments of 16x16.
#define TM 128
#define TN 128
#define KB 32
#define ASTR 40    // LDS row stride (bf16) for A tile: 80B rows (16B-aligned, bank-spread)
#define BSTR 136   // LDS row stride (bf16) for B tile: 272B rows

__global__ __launch_bounds__(256) void gemm_bf16_async(
    const ushort_t* __restrict__ Ahi, const ushort_t* __restrict__ Alo, int lda,
    const ushort_t* __restrict__ Whi, const ushort_t* __restrict__ Wlo, int ldw,
    const float* __restrict__ bias,
    float* __restrict__ C, int ldc,
    int M, int N, int Kpad)
{
  __shared__ unsigned short sAh[TM * ASTR];
  __shared__ unsigned short sAl[TM * ASTR];
  __shared__ unsigned short sBh[KB * BSTR];
  __shared__ unsigned short sBl[KB * BSTR];

  const int tid  = threadIdx.x;
  const int lane = tid & 31;
  const int wave = tid >> 5;
  const int wm   = wave & 3;   // M sub-tile (x32)
  const int wn   = wave >> 2;  // N sub-tile (x64)
  const int m0   = blockIdx.y * TM;
  const int n0   = blockIdx.x * TN;

  const unsigned baseAh = (unsigned)(size_t)&sAh[0];
  const unsigned baseAl = (unsigned)(size_t)&sAl[0];
  const unsigned baseBh = (unsigned)(size_t)&sBh[0];
  const unsigned baseBl = (unsigned)(size_t)&sBl[0];

  v8f acc[2][4];
  #pragma unroll
  for (int i = 0; i < 2; ++i)
    #pragma unroll
    for (int j = 0; j < 4; ++j)
      #pragma unroll
      for (int k = 0; k < 8; ++k) acc[i][j][k] = 0.f;

  for (int k0 = 0; k0 < Kpad; k0 += KB) {
    // --- async stage A tiles (128x32 hi + lo): 512 chunks of 16B each ---
    #pragma unroll
    for (int it = 0; it < 2; ++it) {
      int a   = it * 256 + tid;          // 0..511
      int row = a >> 2;                  // 0..127
      int ch  = a & 3;                   // 4 x 16B per 64B row
      unsigned lds = (unsigned)(row * (ASTR * 2) + ch * 16);
      size_t goff = (size_t)(m0 + row) * lda + k0 + ch * 8;
      async_copy16(baseAh + lds, Ahi + goff);
      async_copy16(baseAl + lds, Alo + goff);
    }
    // --- async stage B tiles (32x128 hi + lo): 512 chunks of 16B each ---
    #pragma unroll
    for (int it = 0; it < 2; ++it) {
      int a   = it * 256 + tid;          // 0..511
      int row = a >> 4;                  // 0..31
      int ch  = a & 15;                  // 16 x 16B per 256B row
      unsigned lds = (unsigned)(row * (BSTR * 2) + ch * 16);
      size_t goff = (size_t)(k0 + row) * ldw + n0 + ch * 8;
      async_copy16(baseBh + lds, Whi + goff);
      async_copy16(baseBl + lds, Wlo + goff);
    }
    wait_async0();
    __syncthreads();

    // --- A fragments: lane(0..15)=row M, lane-half picks K-octets (ISA layout) ---
    const int arow = lane & 15;
    const int j0   = (lane >> 4) << 3;   // 0 or 8
    v16bf ah[2], al[2];
    #pragma unroll
    for (int mi = 0; mi < 2; ++mi) {
      int row = wm * 32 + mi * 16 + arow;
      const unsigned short* ph = &sAh[row * ASTR];
      const unsigned short* pl = &sAl[row * ASTR];
      BFV th, tl;
      #pragma unroll
      for (int t = 0; t < 8; ++t)  { th.u[t] = ph[j0 + t];      tl.u[t] = pl[j0 + t]; }
      #pragma unroll
      for (int t = 8; t < 16; ++t) { th.u[t] = ph[j0 + 8 + t];  tl.u[t] = pl[j0 + 8 + t]; }
      ah[mi] = th.v; al[mi] = tl.v;
    }
    // --- B fragments: lane(0..15)=column N, lane-half picks K-octets (same K order as A) ---
    v16bf bh[4], bl[4];
    #pragma unroll
    for (int ni = 0; ni < 4; ++ni) {
      const unsigned short* ph = &sBh[wn * 64 + ni * 16 + (lane & 15)];
      const unsigned short* pl = &sBl[wn * 64 + ni * 16 + (lane & 15)];
      BFV th, tl;
      #pragma unroll
      for (int t = 0; t < 16; ++t) { int kk = (t < 8) ? (j0 + t) : (j0 + 8 + t); th.u[t] = ph[kk * BSTR]; tl.u[t] = pl[kk * BSTR]; }
      bh[ni] = th.v; bl[ni] = tl.v;
    }

    // --- 3 WMMAs per fragment: Ah*Bh + Ah*Bl + Al*Bh (fp32-class accuracy) ---
    #pragma unroll
    for (int mi = 0; mi < 2; ++mi)
      #pragma unroll
      for (int ni = 0; ni < 4; ++ni) {
        acc[mi][ni] = __builtin_amdgcn_wmma_f32_16x16x32_bf16(
            false, ah[mi], false, bh[ni], (short)0, acc[mi][ni], false, false);
        acc[mi][ni] = __builtin_amdgcn_wmma_f32_16x16x32_bf16(
            false, ah[mi], false, bl[ni], (short)0, acc[mi][ni], false, false);
        acc[mi][ni] = __builtin_amdgcn_wmma_f32_16x16x32_bf16(
            false, al[mi], false, bh[ni], (short)0, acc[mi][ni], false, false);
      }
    __syncthreads();
  }

  // --- epilogue: C layout lane(0..15)=N, lane-half = +8 rows, VGPR r = M ---
  const int cn = lane & 15;
  const int rp = (lane >> 4) << 3;
  #pragma unroll
  for (int mi = 0; mi < 2; ++mi)
    #pragma unroll
    for (int ni = 0; ni < 4; ++ni) {
      int gcol = n0 + wn * 64 + ni * 16 + cn;
      if (gcol >= N) continue;
      float bv = bias ? bias[gcol] : 0.f;
      #pragma unroll
      for (int r = 0; r < 8; ++r) {
        int grow = m0 + wm * 32 + mi * 16 + rp + r;
        if (grow < M) C[(size_t)grow * ldc + gcol] = acc[mi][ni][r] + bv;
      }
    }
}

// ---------------- fp32 -> padded bf16 hi/lo split ----------------
__global__ void split_pad(const float* __restrict__ src, int rows, int cols, int ld_src,
                          int rowsP, int colsP,
                          ushort_t* __restrict__ hi, ushort_t* __restrict__ lo) {
  size_t i = (size_t)blockIdx.x * blockDim.x + threadIdx.x;
  size_t total = (size_t)rowsP * colsP;
  if (i >= total) return;
  int r = (int)(i / colsP), c = (int)(i % colsP);
  float v = (r < rows && c < cols) ? src[(size_t)r * ld_src + c] : 0.f;
  unsigned short h, l; split2(v, h, l);
  hi[i] = h; lo[i] = l;
}

// encoder input assembled directly in split form: [NTPAD x 96] hi/lo
// rows<NG: [X(78)|init_grid(3)|0], NG<=rows<NT: [0(78)|init_mesh(3)|0], rest 0
__global__ void pad_nodes_split(const float* __restrict__ X, const float* __restrict__ ig,
                                const float* __restrict__ im,
                                ushort_t* __restrict__ hi, ushort_t* __restrict__ lo) {
  size_t i = (size_t)blockIdx.x * blockDim.x + threadIdx.x;
  size_t total = (size_t)NTPAD * 96;
  if (i >= total) return;
  int r = (int)(i / 96), c = (int)(i % 96);
  float v = 0.f;
  if (r < NGRID) {
    if (c < 78) v = X[(size_t)r * 78 + c];
    else if (c < 81) v = ig[(size_t)r * 3 + (c - 78)];
  } else if (r < NTOT) {
    if (c >= 78 && c < 81) v = im[(size_t)(r - NGRID) * 3 + (c - 78)];
  }
  unsigned short h, l; split2(v, h, l);
  hi[i] = h; lo[i] = l;
}

// ---------------- graph degree / dinv ----------------
__global__ void deg_init(float* deg, int n) {
  int i = blockIdx.x * blockDim.x + threadIdx.x;
  if (i < n) deg[i] = 1.0f;   // self loop
}
__global__ void deg_edges(float* deg, const int* __restrict__ dst, int E) {
  int i = blockIdx.x * blockDim.x + threadIdx.x;
  if (i < E) atomic_add_f32(&deg[dst[i]], 1.0f);
}
__global__ void deg_rsqrt(float* deg, int n) {
  int i = blockIdx.x * blockDim.x + threadIdx.x;
  if (i < n) deg[i] = rsqrtf(deg[i]);  // becomes dinv
}

// ---------------- GCN aggregation ----------------
__global__ void agg_init(const float* __restrict__ H, const float* __restrict__ dinv,
                         float* __restrict__ out, int rows, int N) {
  size_t i = (size_t)blockIdx.x * blockDim.x + threadIdx.x;
  size_t total = (size_t)rows * N;
  if (i >= total) return;
  int r = (int)(i / N);
  float d = dinv[r];
  out[i] = H[i] * d * d;
}
__global__ void agg_edges(const float* __restrict__ H, const int* __restrict__ src,
                          const int* __restrict__ dst, const float* __restrict__ dinv,
                          float* __restrict__ out, int E, int N) {
  int e = blockIdx.x * (blockDim.x >> 5) + (threadIdx.x >> 5);
  if (e >= E) return;
  int lane = threadIdx.x & 31;
  int s = src[e], d = dst[e];
  float c = dinv[s] * dinv[d];
  const float* h = H + (size_t)s * N;
  float* o = out + (size_t)d * N;
  for (int f = lane; f < N; f += 32) atomic_add_f32(&o[f], h[f] * c);
}

// ---------------- elementwise: x = prelu(x + bias, a) ----------------
__global__ void prelu_bias(float* __restrict__ x, const float* __restrict__ bias,
                           const float* __restrict__ a, int rows, int N) {
  size_t i = (size_t)blockIdx.x * blockDim.x + threadIdx.x;
  size_t total = (size_t)rows * N;
  if (i >= total) return;
  float alpha = a[0];
  int f = (int)(i % N);
  float v = x[i] + bias[f];
  x[i] = (v >= 0.f) ? v : alpha * v;
}

// ---------------- layernorm (wave per row), optional pre-bias, in place ----------------
__global__ void ln_rows(float* __restrict__ X, const float* __restrict__ g,
                        const float* __restrict__ b, const float* __restrict__ bias,
                        int rows, int N) {
  int row = blockIdx.x * (blockDim.x >> 5) + (threadIdx.x >> 5);
  if (row >= rows) return;
  int lane = threadIdx.x & 31;
  float* x = X + (size_t)row * N;
  float s = 0.f, s2 = 0.f;
  for (int f = lane; f < N; f += 32) {
    float v = x[f] + (bias ? bias[f] : 0.f);
    s += v; s2 += v * v;
  }
  for (int off = 16; off; off >>= 1) {
    s  += __shfl_xor(s,  off, 32);
    s2 += __shfl_xor(s2, off, 32);
  }
  float mu  = s / N;
  float var = s2 / N - mu * mu;
  float inv = rsqrtf(var + 1e-5f);
  for (int f = lane; f < N; f += 32) {
    float v = x[f] + (bias ? bias[f] : 0.f);
    x[f] = (v - mu) * inv * g[f] + b[f];
  }
}

// final: out[row,:78] = LN(AGG[row]+bias)*g+b + X[row,:78], rows < NGRID only
__global__ void ln_final(const float* __restrict__ AGG, const float* __restrict__ bias,
                         const float* __restrict__ g, const float* __restrict__ b,
                         const float* __restrict__ X, float* __restrict__ out,
                         int rows, int N) {
  int row = blockIdx.x * (blockDim.x >> 5) + (threadIdx.x >> 5);
  if (row >= rows) return;
  int lane = threadIdx.x & 31;
  const float* x = AGG + (size_t)row * N;
  float s = 0.f, s2 = 0.f;
  for (int f = lane; f < N; f += 32) {
    float v = x[f] + bias[f];
    s += v; s2 += v * v;
  }
  for (int off = 16; off; off >>= 1) {
    s  += __shfl_xor(s,  off, 32);
    s2 += __shfl_xor(s2, off, 32);
  }
  float mu  = s / N;
  float var = s2 / N - mu * mu;
  float inv = rsqrtf(var + 1e-5f);
  for (int f = lane; f < N; f += 32) {
    float v = x[f] + bias[f];
    out[(size_t)row * N + f] = (v - mu) * inv * g[f] + b[f] + X[(size_t)row * N + f];
  }
}

// out[i] = i<NG ? top[i] : bot[i-NG]   (rows of width N)
__global__ void concat_rows(const float* __restrict__ top, const float* __restrict__ bot,
                            float* __restrict__ out, int N) {
  size_t i = (size_t)blockIdx.x * blockDim.x + threadIdx.x;
  size_t total = (size_t)NTOT * N;
  if (i >= total) return;
  int r = (int)(i / N);
  out[i] = (r < NGRID) ? top[i] : bot[i - (size_t)NGRID * N];
}

// ---------------- host orchestration ----------------
static inline int cdiv(int a, int b) { return (a + b - 1) / b; }
static inline int cdivs(size_t a, int b) { return (int)((a + b - 1) / b); }

static void run_gemm(const ushort_t* Ahi, const ushort_t* Alo, int lda,
                     const ushort_t* Whi, const ushort_t* Wlo, int ldw,
                     const float* bias, float* C, int ldc,
                     int M, int N, int Kpad, hipStream_t s) {
  dim3 g(cdiv(N, TN), cdiv(M, TM));
  gemm_bf16_async<<<g, 256, 0, s>>>(Ahi, Alo, lda, Whi, Wlo, ldw, bias, C, ldc, M, N, Kpad);
}

static void run_split(const float* src, int rows, int cols, int ld_src,
                      int rowsP, int colsP, ushort_t* hi, ushort_t* lo, hipStream_t s) {
  size_t total = (size_t)rowsP * colsP;
  split_pad<<<cdivs(total, 256), 256, 0, s>>>(src, rows, cols, ld_src, rowsP, colsP, hi, lo);
}

extern "C" void kernel_launch(void* const* d_in, const int* in_sizes, int n_in,
                              void* d_out, int out_size, void* d_ws, size_t ws_size,
                              hipStream_t stream) {
  const float* X   = (const float*)d_in[0];
  const float* ig  = (const float*)d_in[1];
  const float* im  = (const float*)d_in[2];
  const int* enc_ei  = (const int*)d_in[3];
  const int* proc_ei = (const int*)d_in[4];
  const int* dec_ei  = (const int*)d_in[5];
  auto P32 = [&](int i) { return (const float*)d_in[i]; };
  const int EB = 6, PB = 17, DB = 28;  // +{0:lin_w,1:lin_b,2:ln1_g,3:ln1_b,4:gw1,5:gb1,6:a,7:gw2,8:gb2,9:ln2_g,10:ln2_b}

  // ---- workspace carve (bytes, 256B aligned) ----
  char* wsb = (char*)d_ws;
  size_t off = 0;
  auto carve = [&](size_t bytes) { size_t o = off; off += (bytes + 255) & ~(size_t)255; return o; };
  float* A1 = (float*)(wsb + carve((size_t)NTOT * 256 * 4));
  float* B1 = (float*)(wsb + carve((size_t)NTOT * 256 * 4));
  float* C1 = (float*)(wsb + carve((size_t)NTOT * 256 * 4));
  float* Q1 = (float*)(wsb + carve((size_t)NMESH * 256 * 4));
  float* Q2 = (float*)(wsb + carve((size_t)NMESH * 256 * 4));
  float* Q3 = (float*)(wsb + carve((size_t)NMESH * 256 * 4));
  float* dE = (float*)(wsb + carve((size_t)NTOT * 4));
  float* dP = (float*)(wsb + carve((size_t)NMESH * 4));
  float* dD = (float*)(wsb + carve((size_t)NTOT * 4));
  ushort_t* Ah = (ushort_t*)(wsb + carve((size_t)NTPAD * 256 * 2));  // activation hi
  ushort_t* Al = (ushort_t*)(wsb + carve((size_t)NTPAD * 256 * 2));  // activation lo
  // pre-split weights (padded): [Kpad x Npad]
  ushort_t *W_[9][2]; int wk[9], wn_[9];
  {
    struct { int kp, np; } wd[9] = {
      {96, 256},  {256, 256}, {256, 256},   // enc lin_w(81x256), gw1, gw2
      {256, 256}, {256, 256}, {256, 256},   // proc lin_w, gw1, gw2
      {256, 256}, {256, 256}, {256, 128}};  // dec lin_w, gw1, gw2(256x78)
    for (int i = 0; i < 9; ++i) {
      wk[i] = wd[i].kp; wn_[i] = wd[i].np;
      W_[i][0] = (ushort_t*)(wsb + carve((size_t)wd[i].kp * wd[i].np * 2));
      W_[i][1] = (ushort_t*)(wsb + carve((size_t)wd[i].kp * wd[i].np * 2));
    }
  }
  float* out = (float*)d_out;

  // ---- split all weights once ----
  {
    const float* wsrc[9] = {P32(EB+0), P32(EB+4), P32(EB+7),
                            P32(PB+0), P32(PB+4), P32(PB+7),
                            P32(DB+0), P32(DB+4), P32(DB+7)};
    int wr[9] = {81, 256, 256, 256, 256, 256, 256, 256, 256};
    int wc[9] = {256, 256, 256, 256, 256, 256, 256, 256, 78};
    for (int i = 0; i < 9; ++i)
      run_split(wsrc[i], wr[i], wc[i], wc[i], wk[i], wn_[i], W_[i][0], W_[i][1], stream);
  }

  // ---- degrees -> dinv for the three graphs ----
  deg_init<<<cdiv(NTOT, 256), 256, 0, stream>>>(dE, NTOT);
  deg_edges<<<cdiv(EENC, 256), 256, 0, stream>>>(dE, enc_ei + EENC, EENC);
  deg_rsqrt<<<cdiv(NTOT, 256), 256, 0, stream>>>(dE, NTOT);
  deg_init<<<cdiv(NMESH, 256), 256, 0, stream>>>(dP, NMESH);
  deg_edges<<<cdiv(EPROC, 256), 256, 0, stream>>>(dP, proc_ei + EPROC, EPROC);
  deg_rsqrt<<<cdiv(NMESH, 256), 256, 0, stream>>>(dP, NMESH);
  deg_init<<<cdiv(NTOT, 256), 256, 0, stream>>>(dD, NTOT);
  deg_edges<<<cdiv(EDEC, 256), 256, 0, stream>>>(dD, dec_ei + EDEC, EDEC);
  deg_rsqrt<<<cdiv(NTOT, 256), 256, 0, stream>>>(dD, NTOT);

  const size_t NE = (size_t)NTOT * 256;
  const size_t NEM = (size_t)NMESH * 256;

  // ================= ENCODER (n = NTOT, graph = enc) =================
  pad_nodes_split<<<cdivs((size_t)NTPAD * 96, 256), 256, 0, stream>>>(X, ig, im, Ah, Al);
  run_gemm(Ah, Al, 96, W_[0][0], W_[0][1], 256, P32(EB+1), A1, 256, NTOT, 256, 96, stream);
  ln_rows<<<cdiv(NTOT, 8), 256, 0, stream>>>(A1, P32(EB+2), P32(EB+3), nullptr, NTOT, 256);
  run_split(A1, NTOT, 256, 256, NTPAD, 256, Ah, Al, stream);
  run_gemm(Ah, Al, 256, W_[1][0], W_[1][1], 256, nullptr, B1, 256, NTOT, 256, 256, stream);
  agg_init<<<cdivs(NE, 256), 256, 0, stream>>>(B1, dE, C1, NTOT, 256);
  agg_edges<<<cdiv(EENC, 8), 256, 0, stream>>>(B1, enc_ei, enc_ei + EENC, dE, C1, EENC, 256);
  prelu_bias<<<cdivs(NE, 256), 256, 0, stream>>>(C1, P32(EB+5), P32(EB+6), NTOT, 256);
  run_split(C1, NTOT, 256, 256, NTPAD, 256, Ah, Al, stream);
  run_gemm(Ah, Al, 256, W_[2][0], W_[2][1], 256, nullptr, B1, 256, NTOT, 256, 256, stream);
  agg_init<<<cdivs(NE, 256), 256, 0, stream>>>(B1, dE, A1, NTOT, 256);
  agg_edges<<<cdiv(EENC, 8), 256, 0, stream>>>(B1, enc_ei, enc_ei + EENC, dE, A1, EENC, 256);
  ln_rows<<<cdiv(NTOT, 8), 256, 0, stream>>>(A1, P32(EB+9), P32(EB+10), P32(EB+8), NTOT, 256);
  // A1 = ENC [NTOT,256]

  // ================= PROCESSOR (n = NMESH, on ENC[NG:]) =================
  const float* ENCm = A1 + (size_t)NGRID * 256;
  run_split(ENCm, NMESH, 256, 256, NMPAD, 256, Ah, Al, stream);
  run_gemm(Ah, Al, 256, W_[3][0], W_[3][1], 256, P32(PB+1), Q1, 256, NMESH, 256, 256, stream);
  ln_rows<<<cdiv(NMESH, 8), 256, 0, stream>>>(Q1, P32(PB+2), P32(PB+3), nullptr, NMESH, 256);
  run_split(Q1, NMESH, 256, 256, NMPAD, 256, Ah, Al, stream);
  run_gemm(Ah, Al, 256, W_[4][0], W_[4][1], 256, nullptr, Q2, 256, NMESH, 256, 256, stream);
  agg_init<<<cdivs(NEM, 256), 256, 0, stream>>>(Q2, dP, Q3, NMESH, 256);
  agg_edges<<<cdiv(EPROC, 8), 256, 0, stream>>>(Q2, proc_ei, proc_ei + EPROC, dP, Q3, EPROC, 256);
  prelu_bias<<<cdivs(NEM, 256), 256, 0, stream>>>(Q3, P32(PB+5), P32(PB+6), NMESH, 256);
  run_split(Q3, NMESH, 256, 256, NMPAD, 256, Ah, Al, stream);
  run_gemm(Ah, Al, 256, W_[5][0], W_[5][1], 256, nullptr, Q2, 256, NMESH, 256, 256, stream);
  agg_init<<<cdivs(NEM, 256), 256, 0, stream>>>(Q2, dP, Q1, NMESH, 256);
  agg_edges<<<cdiv(EPROC, 8), 256, 0, stream>>>(Q2, proc_ei, proc_ei + EPROC, dP, Q1, EPROC, 256);
  ln_rows<<<cdiv(NMESH, 8), 256, 0, stream>>>(Q1, P32(PB+9), P32(PB+10), P32(PB+8), NMESH, 256);
  // Q1 = MESH [NMESH,256]

  // ================= DECODER (n = NTOT, graph = dec) =================
  concat_rows<<<cdivs(NE, 256), 256, 0, stream>>>(A1, Q1, B1, 256);  // B1 = [ENC[:NG]; MESH]
  run_split(B1, NTOT, 256, 256, NTPAD, 256, Ah, Al, stream);
  run_gemm(Ah, Al, 256, W_[6][0], W_[6][1], 256, P32(DB+1), C1, 256, NTOT, 256, 256, stream);
  ln_rows<<<cdiv(NTOT, 8), 256, 0, stream>>>(C1, P32(DB+2), P32(DB+3), nullptr, NTOT, 256);
  run_split(C1, NTOT, 256, 256, NTPAD, 256, Ah, Al, stream);
  run_gemm(Ah, Al, 256, W_[7][0], W_[7][1], 256, nullptr, A1, 256, NTOT, 256, 256, stream);
  agg_init<<<cdivs(NE, 256), 256, 0, stream>>>(A1, dD, B1, NTOT, 256);
  agg_edges<<<cdiv(EDEC, 8), 256, 0, stream>>>(A1, dec_ei, dec_ei + EDEC, dD, B1, EDEC, 256);
  prelu_bias<<<cdivs(NE, 256), 256, 0, stream>>>(B1, P32(DB+5), P32(DB+6), NTOT, 256);
  run_split(B1, NTOT, 256, 256, NTPAD, 256, Ah, Al, stream);
  run_gemm(Ah, Al, 256, W_[8][0], W_[8][1], 128, nullptr, A1, 78, NTOT, 78, 256, stream);
  agg_init<<<cdivs((size_t)NTOT * 78, 256), 256, 0, stream>>>(A1, dD, C1, NTOT, 78);
  agg_edges<<<cdiv(EDEC, 8), 256, 0, stream>>>(A1, dec_ei, dec_ei + EDEC, dD, C1, EDEC, 78);
  ln_final<<<cdiv(NGRID, 8), 256, 0, stream>>>(C1, P32(DB+8), P32(DB+9), P32(DB+10),
                                               X, out, NGRID, 78);
  (void)in_sizes; (void)n_in; (void)out_size; (void)ws_size;
}
